// Lobe_28321014350246
// MI455X (gfx1250) — matrix operand order repaired and checked by reading, compiled, not run on hardware
//
#include <hip/hip_runtime.h>
#include <math.h>

// ---------------------------------------------------------------------------
// Problem constants (match reference)
// ---------------------------------------------------------------------------
constexpr int B_   = 32;
constexpr int N_   = 512;
constexpr int CIN_ = 256;
constexpr int H_   = 256;
constexpr int NH_  = 8;
constexpr int DH_  = 32;    // H_/NH_
constexpr int DFF_ = 512;
constexpr int OUT_ = 2;
constexpr int NLAYERS_ = 4;
constexpr int R_   = B_ * N_;          // 16384 rows
constexpr float RCUT_ = 8.0f;

// ---------------------------------------------------------------------------
// WMMA types (CDNA5 gfx1250, wave32)
// ---------------------------------------------------------------------------
typedef __attribute__((ext_vector_type(16))) __bf16 v16bf;
typedef __attribute__((ext_vector_type(8)))  float  v8f;

union Frag16 { v16bf v; unsigned u[8]; __bf16 e[16]; };
union FAcc   { v8f v; float f[8]; };

__device__ __forceinline__ v8f wmma_bf16(const Frag16& a, const Frag16& b, v8f c) {
    return __builtin_amdgcn_wmma_f32_16x16x32_bf16(false, a.v, false, b.v,
                                                   (short)0, c, false, false);
}

// CDNA5 async global->LDS copy (16B per lane), tracked by ASYNCcnt.
__device__ __forceinline__ void async_copy_b128(unsigned lds_addr, const void* gaddr) {
    asm volatile("global_load_async_to_lds_b128 %0, %1, off"
                 :: "v"(lds_addr), "v"(gaddr) : "memory");
}
__device__ __forceinline__ void wait_async0() {
#if defined(__has_builtin) && __has_builtin(__builtin_amdgcn_s_wait_asynccnt)
    __builtin_amdgcn_s_wait_asynccnt(0);
#else
    asm volatile("s_wait_asynccnt 0x0" ::: "memory");
#endif
}
__device__ __forceinline__ void sched_fence() {
#if defined(__has_builtin) && __has_builtin(__builtin_amdgcn_sched_barrier)
    __builtin_amdgcn_sched_barrier(0);
#endif
}

// ---------------------------------------------------------------------------
// Generic batched GEMM:  C[z] = act(A[z] * B[z] + bias) + Cadd
//   A  : bf16 row-major [M][K],  lda elements between rows
//   BT : bf16 row-major [N][K] (i.e. B transposed), ldb elements between rows
// Both tiles staged by async global->LDS copies (no VGPR round trip, no
// transpose scatter). Double-buffered, one barrier per K-step, 4 WMMAs/step.
// Outputs: C (f32), Cbf (bf16, same layout), CbfT (bf16 transposed [N][M],
// ldct = M) - each optional.
// Requirements: M%64==0, N%128==0, K%32==0, lda/ldb mult of 8, 16B-aligned.
// ---------------------------------------------------------------------------
#define GBM 64
#define GBN 128
#define GBK 32

__global__ __launch_bounds__(256)
void gemm_bf16_kernel(const __bf16* __restrict__ A, const __bf16* __restrict__ BT,
                      float* __restrict__ C, __bf16* __restrict__ Cbf,
                      __bf16* __restrict__ CbfT, int ldct,
                      const float* __restrict__ bias, const float* __restrict__ Cadd,
                      int M, int Ncols, int K, int lda, int ldb, int ldc,
                      long long sA, long long sB, long long sC, int act)
{
    (void)Ncols;
    __shared__ __bf16 As[2][GBM][GBK + 8];   // [buf][m][k]
    __shared__ __bf16 Bs[2][GBN][GBK + 8];   // [buf][n][k]

    const int z = blockIdx.z;
    A  += (long long)z * sA;
    BT += (long long)z * sB;
    const long long coff = (long long)z * sC;

    const int m0 = blockIdx.y * GBM;
    const int n0 = blockIdx.x * GBN;
    const int t    = threadIdx.x;
    const int wave = t >> 5;
    const int lane = t & 31;
    const int mt = wave & 3;        // m-tile 0..3
    const int np = wave >> 2;       // 0..1, 64 cols each (4 n-tiles)
    const int nl = lane & 15;
    const int hf = lane >> 4;

    FAcc acc[4];
#pragma unroll
    for (int q = 0; q < 4; q++)
#pragma unroll
        for (int i = 0; i < 8; i++) acc[q].f[i] = 0.f;

    const int ar = t >> 2, ac = (t & 3) * 8;   // A tile 64x32 : 8 elems/thread
    const int bn = t >> 1, bk = (t & 1) * 16;  // B tile 128x32: 16 elems/thread

    auto asyncA = [&](int kk, int buf) {
        const unsigned laddr = (unsigned)(uintptr_t)&As[buf][ar][ac];
        async_copy_b128(laddr, A + (long long)(m0 + ar) * lda + (kk + ac));
    };
    auto asyncB = [&](int kk, int buf) {
        const unsigned laddr = (unsigned)(uintptr_t)&Bs[buf][bn][bk];
        const __bf16* gp = BT + (long long)(n0 + bn) * ldb + (kk + bk);
        async_copy_b128(laddr,      gp);
        async_copy_b128(laddr + 16, gp + 8);
    };

    const int nK = K / GBK;

    // prologue: stage tile 0 into buffer 0
    asyncA(0, 0);
    asyncB(0, 0);
    wait_async0();
    __syncthreads();

    int buf = 0;
    for (int ki = 0; ki < nK; ++ki) {
        const bool more = (ki + 1 < nK);
        if (more) {                       // issue next tile early (latency hidden)
            asyncA((ki + 1) * GBK, buf ^ 1);
            asyncB((ki + 1) * GBK, buf ^ 1);
        }

        // preload ALL fragments, then fire 4 WMMAs back-to-back
        Frag16 fa;
        Frag16 fb[4];
#pragma unroll
        for (int ep = 0; ep < 8; ep++) {
            const int e  = 2 * ep;
            const int ka = (e < 8 ? e : e + 8) + hf * 8;   // A-frag K map
            fa.u[ep] = *(const unsigned*)&As[buf][mt * 16 + nl][ka];
        }
#pragma unroll
        for (int q = 0; q < 4; q++) {
#pragma unroll
            for (int ep = 0; ep < 8; ep++) {
                const int kb = (2 * ep < 8 ? 2 * ep : 2 * ep + 8) + hf * 8;           // B-frag K map
                fb[q].u[ep] = *(const unsigned*)&Bs[buf][np * 64 + q * 16 + nl][kb];
            }
        }
        sched_fence();   // keep all fragments live: one dscnt wait, 4 WMMAs
#pragma unroll
        for (int q = 0; q < 4; q++)
            acc[q].v = wmma_bf16(fa, fb[q], acc[q].v);

        if (more) {
            wait_async0();
            __syncthreads();
            buf ^= 1;
        }
    }

    // ---- epilogue: bias/act first, then each optional output under ONE branch ----
    float outv[4][8];
#pragma unroll
    for (int q = 0; q < 4; q++) {
        const int col = n0 + np * 64 + q * 16 + nl;
        const float bv = bias ? bias[col] : 0.f;
#pragma unroll
        for (int r = 0; r < 8; r++) {
            float vv = acc[q].f[r] + bv;
            if (act == 1)      vv = vv > 0.f ? vv : 0.f;           // relu
            else if (act == 2) vv = vv / (1.f + __expf(-vv));      // silu
            outv[q][r] = vv;
        }
    }
    if (Cadd) {
#pragma unroll
        for (int q = 0; q < 4; q++) {
            const int col = n0 + np * 64 + q * 16 + nl;
#pragma unroll
            for (int r = 0; r < 8; r++) {
                const int row = m0 + mt * 16 + r + hf * 8;
                outv[q][r] += Cadd[coff + (long long)row * ldc + col];
            }
        }
    }
    if (C) {
#pragma unroll
        for (int q = 0; q < 4; q++) {
            const int col = n0 + np * 64 + q * 16 + nl;
#pragma unroll
            for (int r = 0; r < 8; r++) {
                const int row = m0 + mt * 16 + r + hf * 8;
                C[coff + (long long)row * ldc + col] = outv[q][r];
            }
        }
    }
    if (Cbf) {
#pragma unroll
        for (int q = 0; q < 4; q++) {
            const int col = n0 + np * 64 + q * 16 + nl;
#pragma unroll
            for (int r = 0; r < 8; r++) {
                const int row = m0 + mt * 16 + r + hf * 8;
                Cbf[coff + (long long)row * ldc + col] = (__bf16)outv[q][r];
            }
        }
    }
    if (CbfT) {  // transposed bf16 out: lane writes 8 contiguous rows = one b128
        const int row0 = m0 + mt * 16 + hf * 8;
#pragma unroll
        for (int q = 0; q < 4; q++) {
            const int col = n0 + np * 64 + q * 16 + nl;
            union { __bf16 e[8]; uint4 q4; } pk;
#pragma unroll
            for (int r = 0; r < 8; r++) pk.e[r] = (__bf16)outv[q][r];
            *(uint4*)&CbfT[(long long)col * ldct + row0] = pk.q4;
        }
    }
}

// ---------------------------------------------------------------------------
// Flash attention: per (b,h), rows in blocks of 128 (8 waves x 16 rows).
// DH=32 -> scores tile = one WMMA; online softmax; P*V via two WMMAs.
// Double-buffered K/V staging, one barrier per 32-key chunk.
// ---------------------------------------------------------------------------
__global__ __launch_bounds__(256)
void attn_kernel(const __bf16* __restrict__ Qb, const __bf16* __restrict__ Kb,
                 const __bf16* __restrict__ Vb, float* __restrict__ Out)
{
    const int b = blockIdx.z, h = blockIdx.y;
    const int t = threadIdx.x, wave = t >> 5, lane = t & 31;
    const int nl = lane & 15, hf = lane >> 4;
    const int r0 = blockIdx.x * 128 + wave * 16;
    const float SCALE = 0.17677669529663687f;   // 1/sqrt(32)

    __shared__ __bf16 Ks[2][32][40];     // [buf][j][d]
    __shared__ __bf16 Vt[2][32][40];     // [buf][d][j]  (transposed)
    __shared__ __bf16 Ps[8][16][40];     // per-wave P tile [m][j]

    const int jj = t >> 3;               // 0..31 staging row
    const int dd = (t & 7) * 4;          // 0..28 staging col group

    auto loadKV = [&](int j0, uint2& kq, uint2& vq) {
        const long long base = ((long long)(b * N_ + j0 + jj)) * H_ + h * DH_ + dd;
        kq = *(const uint2*)(Kb + base);
        vq = *(const uint2*)(Vb + base);
    };
    auto storeKV = [&](int buf, const uint2& kq, const uint2& vq) {
        unsigned* kd = (unsigned*)&Ks[buf][jj][dd];
        kd[0] = kq.x; kd[1] = kq.y;
        union { uint2 q; __bf16 e[4]; } uv; uv.q = vq;
#pragma unroll
        for (int i = 0; i < 4; i++) Vt[buf][dd + i][jj] = uv.e[i];
    };

    // Q fragment (A 16x32): rows r0..r0+15, K-dim = head dims
    Frag16 fq;
    {
        const __bf16* qp = Qb + ((long long)(b * N_ + r0 + nl)) * H_ + h * DH_;
#pragma unroll
        for (int ep = 0; ep < 8; ep++) {
            const int e = 2 * ep;
            const int k = (e < 8 ? e : e + 8) + hf * 8;
            fq.u[ep] = *(const unsigned*)(qp + k);
        }
    }

    FAcc o0, o1;
    float mi[8], li[8];
#pragma unroll
    for (int r = 0; r < 8; r++) { o0.f[r] = 0.f; o1.f[r] = 0.f; mi[r] = -3.0e38f; li[r] = 0.f; }

    uint2 kq, vq;
    loadKV(0, kq, vq);
    storeKV(0, kq, vq);
    __syncthreads();

    int buf = 0;
    for (int j0 = 0; j0 < N_; j0 += 32) {
        const bool more = (j0 + 32 < N_);
        if (more) loadKV(j0 + 32, kq, vq);

        // scores: S = Q * K^T  (two 16x16 tiles covering 32 keys)
        Frag16 fk0, fk1;
#pragma unroll
        for (int ep = 0; ep < 8; ep++) {
            const int k = (2 * ep < 8 ? 2 * ep : 2 * ep + 8) + hf * 8;
            fk0.u[ep] = *(const unsigned*)&Ks[buf][     nl][k];
            fk1.u[ep] = *(const unsigned*)&Ks[buf][16 + nl][k];
        }
        FAcc s0, s1;
#pragma unroll
        for (int r = 0; r < 8; r++) { s0.f[r] = 0.f; s1.f[r] = 0.f; }
        s0.v = wmma_bf16(fq, fk0, s0.v);
        s1.v = wmma_bf16(fq, fk1, s1.v);

        // V fragments depend only on the staged buffer -> issue early so the
        // LDS traffic overlaps the softmax VALU work below.
        Frag16 fv0, fv1;
#pragma unroll
        for (int ep = 0; ep < 8; ep++) {
            const int k = (2 * ep < 8 ? 2 * ep : 2 * ep + 8) + hf * 8;
            fv0.u[ep] = *(const unsigned*)&Vt[buf][     nl][k];
            fv1.u[ep] = *(const unsigned*)&Vt[buf][16 + nl][k];
        }

        // online softmax per row (row r+hf*8 lives in vgpr slot r of each half)
#pragma unroll
        for (int r = 0; r < 8; r++) {
            float a0 = s0.f[r] * SCALE, a1 = s1.f[r] * SCALE;
            float mx = fmaxf(a0, a1);
            mx = fmaxf(mx, __shfl_xor(mx, 1));
            mx = fmaxf(mx, __shfl_xor(mx, 2));
            mx = fmaxf(mx, __shfl_xor(mx, 4));
            mx = fmaxf(mx, __shfl_xor(mx, 8));
            const float mnew = fmaxf(mi[r], mx);
            const float sc = __expf(mi[r] - mnew);
            mi[r] = mnew;
            o0.f[r] *= sc; o1.f[r] *= sc; li[r] *= sc;
            const float p0 = __expf(a0 - mnew);
            const float p1 = __expf(a1 - mnew);
            float rs = p0 + p1;
            rs += __shfl_xor(rs, 1);
            rs += __shfl_xor(rs, 2);
            rs += __shfl_xor(rs, 4);
            rs += __shfl_xor(rs, 8);
            li[r] += rs;
            // C-layout -> A-layout via wave-private LDS (element (m=r+hf*8, j))
            Ps[wave][r + hf * 8][     nl] = (__bf16)p0;
            Ps[wave][r + hf * 8][16 + nl] = (__bf16)p1;
        }

        // P as A-fragment (16x32 over keys)
        Frag16 fp;
#pragma unroll
        for (int ep = 0; ep < 8; ep++) {
            const int e = 2 * ep;
            const int k = (e < 8 ? e : e + 8) + hf * 8;
            fp.u[ep] = *(const unsigned*)&Ps[wave][nl][k];
        }
        o0.v = wmma_bf16(fp, fv0, o0.v);
        o1.v = wmma_bf16(fp, fv1, o1.v);

        if (more) {
            storeKV(buf ^ 1, kq, vq);
            __syncthreads();
            buf ^= 1;
        }
    }

#pragma unroll
    for (int r = 0; r < 8; r++) {
        const int row = r0 + r + hf * 8;
        const float inv = 1.f / li[r];
        const long long base = ((long long)(b * N_ + row)) * H_ + h * DH_;
        Out[base +      nl] = o0.f[r] * inv;
        Out[base + 16 + nl] = o1.f[r] * inv;
    }
}

// ---------------------------------------------------------------------------
// Elementwise / fusion kernels
// ---------------------------------------------------------------------------
__global__ void cast_kernel(const float* __restrict__ in, __bf16* __restrict__ out, int n) {
    const int i = blockIdx.x * 256 + threadIdx.x;
    if (i < n) out[i] = (__bf16)in[i];
}

// f32 [rows][cols] -> bf16 transposed [cols][rows]
__global__ void transpose_bf_kernel(const float* __restrict__ in, __bf16* __restrict__ out,
                                    int rows, int cols) {
    const int i = blockIdx.x * 256 + threadIdx.x;   // over rows*cols (multiple of 256)
    const int c = i / rows, r = i - c * rows;
    out[i] = (__bf16)in[(long long)r * cols + c];   // out[c*rows + r]
}

// graph_features[:, :, 1:4, :] -> bf16 (R,3,CIN)
__global__ void extract_v_kernel(const float* __restrict__ gf, __bf16* __restrict__ out) {
    const int i = blockIdx.x * 256 + threadIdx.x;     // i < R*3*CIN
    const int k = i & 255;
    const int rc = i >> 8;
    const int c = rc % 3;
    const int r = rc / 3;
    out[i] = (__bf16)gf[(((long long)r * 4) + 1 + c) * CIN_ + k];
}

// concat([x, ||vec1||_c]) as bf16 (R, 2H); vec1 rows are (bj,c) interleaved
__global__ void concat_kernel(const float* __restrict__ gf, const float* __restrict__ vec1,
                              __bf16* __restrict__ out) {
    const int r = blockIdx.x, t = threadIdx.x;
    out[(long long)r * 512 + t] = (__bf16)gf[((long long)r * 4) * CIN_ + t];
    float s = 1e-8f;
#pragma unroll
    for (int c = 0; c < 3; c++) {
        const float d = vec1[((long long)r * 3 + c) * H_ + t];
        s += d * d;
    }
    out[(long long)r * 512 + 256 + t] = (__bf16)sqrtf(s);
}

// x = hout[:, :H]; v = vec2 * gate. v stored channel-major: (3, B*N, H)
__global__ void gate_kernel(const float* __restrict__ hout, const float* __restrict__ vec2,
                            float* __restrict__ v, __bf16* __restrict__ vbf,
                            float* __restrict__ x, __bf16* __restrict__ xbf) {
    const int r = blockIdx.x, t = threadIdx.x;
    const float xv = hout[(long long)r * 512 + t];
    x  [(long long)r * H_ + t] = xv;
    xbf[(long long)r * H_ + t] = (__bf16)xv;
    const float g = hout[(long long)r * 512 + 256 + t];
#pragma unroll
    for (int c = 0; c < 3; c++) {
        const long long src = ((long long)r * 3 + c) * H_ + t;      // vec2 (bj,c)
        const long long dst = ((long long)c * R_ + r) * H_ + t;     // v (c,bj)
        const float vv = vec2[src] * g;
        v[dst] = vv;
        vbf[dst] = (__bf16)vv;
    }
}

// cutoff weights w (B,N,N) and w*dirn (B,3,N,N), bf16
__global__ void buildw_kernel(const float* __restrict__ coords,
                              __bf16* __restrict__ w, __bf16* __restrict__ wdir) {
    const int j = blockIdx.x * 256 + threadIdx.x;
    const int i = blockIdx.y;
    const int b = blockIdx.z;
    const float* ci = coords + ((long long)b * N_ + i) * 3;
    const float* cj = coords + ((long long)b * N_ + j) * 3;
    const float dx = ci[0] - cj[0], dy = ci[1] - cj[1], dz = ci[2] - cj[2];
    const float dist = sqrtf(dx * dx + dy * dy + dz * dz + 1e-12f);
    const bool m = (dist < RCUT_) && (i != j);
    const float wv = m ? 0.5f * (cosf(3.14159265358979f * dist / RCUT_) + 1.0f) : 0.0f;
    w[((long long)b * N_ + i) * N_ + j] = (__bf16)wv;
    const float q = wv / dist;
    const long long nn = (long long)N_ * N_;
    const long long base = (long long)i * N_ + j;
    wdir[((long long)(b * 3 + 0)) * nn + base] = (__bf16)(q * dx);
    wdir[((long long)(b * 3 + 1)) * nn + base] = (__bf16)(q * dy);
    wdir[((long long)(b * 3 + 2)) * nn + base] = (__bf16)(q * dz);
}

// ||v + agg_v||_c -> bf16; resid = x + agg_x + PE  (v/aggv channel-major)
__global__ void postgc_kernel(const float* __restrict__ v, const float* __restrict__ aggv,
                              const float* __restrict__ x, const float* __restrict__ aggx,
                              __bf16* __restrict__ normv, float* __restrict__ resid) {
    const int r = blockIdx.x, t = threadIdx.x;
    const int n = r % N_;
    float s = 1e-8f;
#pragma unroll
    for (int c = 0; c < 3; c++) {
        const long long idx = ((long long)c * R_ + r) * H_ + t;
        const float vv = v[idx] + aggv[idx];
        s += vv * vv;
    }
    normv[(long long)r * H_ + t] = (__bf16)sqrtf(s);
    const int kp = t >> 1;
    const float ang = (float)n * __expf(-(float)(2 * kp) * (logf(10000.f) / (float)H_));
    const float pe = (t & 1) ? cosf(ang) : sinf(ang);
    resid[(long long)r * H_ + t] = x[(long long)r * H_ + t] + aggx[(long long)r * H_ + t] + pe;
}

// LayerNorm over H=256 (one block per row)
__global__ __launch_bounds__(256)
void ln_kernel(const float* __restrict__ in, const float* __restrict__ g,
               const float* __restrict__ bb, float* __restrict__ outf,
               __bf16* __restrict__ outbf) {
    __shared__ float red[256];
    const int r = blockIdx.x, t = threadIdx.x;
    const float v = in[(long long)r * H_ + t];
    red[t] = v; __syncthreads();
    for (int s = 128; s > 0; s >>= 1) { if (t < s) red[t] += red[t + s]; __syncthreads(); }
    const float mean = red[0] * (1.0f / H_);
    __syncthreads();
    const float d = v - mean;
    red[t] = d * d; __syncthreads();
    for (int s = 128; s > 0; s >>= 1) { if (t < s) red[t] += red[t + s]; __syncthreads(); }
    const float var = red[0] * (1.0f / H_);
    const float y = d * rsqrtf(var + 1e-5f) * g[t] + bb[t];
    outf [(long long)r * H_ + t] = y;
    outbf[(long long)r * H_ + t] = (__bf16)y;
}

__global__ void pool_kernel(const float* __restrict__ x, float* __restrict__ pooled) {
    const int b = blockIdx.x, t = threadIdx.x;
    float s = 0.f;
    for (int n = 0; n < N_; n++) s += x[((long long)(b * N_ + n)) * H_ + t];
    pooled[b * H_ + t] = s * (1.0f / N_);
}

// tiny head MLP: 256->256(relu)->256(relu)->2, one block per batch row
__global__ __launch_bounds__(256)
void head_kernel(const float* __restrict__ pooled,
                 const float* __restrict__ w1, const float* __restrict__ b1,
                 const float* __restrict__ w2, const float* __restrict__ b2,
                 const float* __restrict__ w3, const float* __restrict__ b3,
                 float* __restrict__ out) {
    __shared__ float si[256], sh[256], red[256];
    const int b = blockIdx.x, t = threadIdx.x;
    si[t] = pooled[b * 256 + t]; __syncthreads();
    float a = b1[t];
    for (int k = 0; k < 256; k++) a += si[k] * w1[k * 256 + t];
    a = a > 0.f ? a : 0.f;
    sh[t] = a; __syncthreads();
    si[t] = sh[t]; __syncthreads();
    float a2 = b2[t];
    for (int k = 0; k < 256; k++) a2 += si[k] * w2[k * 256 + t];
    a2 = a2 > 0.f ? a2 : 0.f;
#pragma unroll
    for (int o = 0; o < OUT_; o++) {
        red[t] = a2 * w3[t * OUT_ + o]; __syncthreads();
        for (int s = 128; s > 0; s >>= 1) { if (t < s) red[t] += red[t + s]; __syncthreads(); }
        if (t == 0) out[b * OUT_ + o] = red[0] + b3[o];
        __syncthreads();
    }
}

// ---------------------------------------------------------------------------
// Host side
// ---------------------------------------------------------------------------
static inline void gemm(hipStream_t s, const __bf16* A, const __bf16* BT,
                        float* C, __bf16* Cbf, __bf16* CbfT, int ldct,
                        const float* bias, const float* Cadd,
                        int M, int Nc, int K, int lda, int ldb, int ldc,
                        long long sA, long long sB, long long sC, int batches, int act) {
    dim3 g(Nc / GBN, M / GBM, batches), blk(256);
    gemm_bf16_kernel<<<g, blk, 0, s>>>(A, BT, C, Cbf, CbfT, ldct, bias, Cadd,
                                       M, Nc, K, lda, ldb, ldc, sA, sB, sC, act);
}

extern "C" void kernel_launch(void* const* d_in, const int* in_sizes, int n_in,
                              void* d_out, int out_size, void* d_ws, size_t ws_size,
                              hipStream_t stream) {
    (void)in_sizes; (void)n_in; (void)out_size; (void)ws_size;

    // Input order: leaves in dict-insertion order.
    const float* gf     = (const float*)d_in[0];   // (B,N,4,CIN)
    const float* coords = (const float*)d_in[1];   // (B,N,3)
    const int P0 = 2;                               // params top-level: 2..17
    auto pf = [&](int i) -> const float* { return (const float*)d_in[i]; };
    const int LBASE = P0 + 16;                      // layer leaves: 16 each

    // -------- workspace arena --------
    char* wsb = (char*)d_ws;
    size_t off = 0;
    auto alloc = [&](size_t bytes) -> char* {
        off = (off + 255) & ~(size_t)255;
        char* p = wsb + off; off += bytes; return p;
    };
    // weight -> bf16 TRANSPOSED [N][K] for direct async B staging
    auto mkbfT = [&](int idx, int rows /*K*/, int cols /*N*/) -> __bf16* {
        const size_t n = (size_t)rows * cols;
        __bf16* p = (__bf16*)alloc(n * 2);
        transpose_bf_kernel<<<dim3((unsigned)(n / 256)), dim3(256), 0, stream>>>(
            pf(idx), p, rows, cols);
        return p;
    };

    __bf16* WT_es_wv1 = mkbfT(P0 + 0, CIN_, H_);
    __bf16* WT_es_wv2 = mkbfT(P0 + 1, CIN_, H_);
    __bf16* WT_es_wu1 = mkbfT(P0 + 2, CIN_ + H_, H_);
    __bf16* WT_es_wu2 = mkbfT(P0 + 4, H_, 2 * H_);
    __bf16* WT_gc_wx  = mkbfT(P0 + 6, H_, H_);
    __bf16* WT_gc_wd  = mkbfT(P0 + 7, H_, H_);
    __bf16* WT_gc_wv  = mkbfT(P0 + 8, H_, H_);
    __bf16* WT_gc_wn  = mkbfT(P0 + 9, H_, H_);
    struct LayerW { __bf16 *wq, *wk, *wv, *wo, *w1, *w2; } LW[NLAYERS_];
    for (int l = 0; l < NLAYERS_; l++) {
        const int base = LBASE + l * 16;
        LW[l].wq = mkbfT(base + 0, H_, H_);
        LW[l].wk = mkbfT(base + 2, H_, H_);
        LW[l].wv = mkbfT(base + 4, H_, H_);
        LW[l].wo = mkbfT(base + 6, H_, H_);
        LW[l].w1 = mkbfT(base + 8, H_, DFF_);
        LW[l].w2 = mkbfT(base + 10, DFF_, H_);
    }

    // activation buffers (with exact-size lifetime aliasing)
    const size_t R3H = (size_t)R_ * 3 * H_;
    const size_t RH  = (size_t)R_ * H_;
    const long long NN = (long long)N_ * N_;

    __bf16* vbf    = (__bf16*)alloc(R3H * 2);        // graph v bf16; later: vvT
    __bf16* vvT    = vbf;                            // [H][3*R_] transposed
    float*  vec1   = (float*)alloc(R3H * 4);         // later: aggv (channel-major)
    float*  aggv   = vec1;
    float*  vec2   = (float*)alloc(R3H * 4);         // later: wdir (3*B*N*N bf16)
    __bf16* wdir   = (__bf16*)vec2;
    __bf16* cat_bf = (__bf16*)alloc((size_t)R_ * 512 * 2); // later: w_bf (B*N*N bf16)
    __bf16* w_bf   = cat_bf;
    __bf16* hmidbf = (__bf16*)alloc(RH * 2);         // later: dxT [H][R_]
    __bf16* dxT    = hmidbf;
    float*  hout   = (float*)alloc((size_t)R_ * 512 * 4);  // later: aggx + resid
    float*  aggx   = hout;
    float*  resid  = hout + RH;
    float*  xf     = (float*)alloc(RH * 4);
    __bf16* xbf    = (__bf16*)alloc(RH * 2);
    float*  vfull  = (float*)alloc(R3H * 4);         // channel-major (3,B*N,H)
    __bf16* vfbf   = (__bf16*)alloc(R3H * 2);        // channel-major
    __bf16* ddT    = (__bf16*)alloc(RH * 2);         // [H][R_]
    __bf16* normv  = (__bf16*)alloc(RH * 2);         // later: qb
    __bf16* qb     = normv;
    __bf16* kb     = (__bf16*)alloc(RH * 2);
    __bf16* vb2    = (__bf16*)alloc(RH * 2);
    float*  attf   = (float*)alloc(RH * 4);
    __bf16* attbf  = (__bf16*)alloc(RH * 2);
    float*  projf  = (float*)alloc(RH * 4);
    float*  x1f    = (float*)alloc(RH * 4);
    __bf16* x1bf   = (__bf16*)alloc(RH * 2);
    __bf16* midbf  = (__bf16*)alloc((size_t)R_ * DFF_ * 2);
    float*  pooled = (float*)alloc((size_t)B_ * H_ * 4);

    // ---- Stage A: equivariant embedding ----
    extract_v_kernel<<<dim3((unsigned)(R3H / 256)), 256, 0, stream>>>(gf, vbf);
    gemm(stream, vbf, WT_es_wv1, vec1, nullptr, nullptr, 0, nullptr, nullptr,
         R_ * 3, H_, CIN_, CIN_, CIN_, H_, 0, 0, 0, 1, 0);
    gemm(stream, vbf, WT_es_wv2, vec2, nullptr, nullptr, 0, nullptr, nullptr,
         R_ * 3, H_, CIN_, CIN_, CIN_, H_, 0, 0, 0, 1, 0);
    concat_kernel<<<R_, 256, 0, stream>>>(gf, vec1, cat_bf);
    gemm(stream, cat_bf, WT_es_wu1, nullptr, hmidbf, nullptr, 0, pf(P0 + 3), nullptr,
         R_, H_, 512, 512, 512, H_, 0, 0, 0, 1, 2 /*silu*/);
    gemm(stream, hmidbf, WT_es_wu2, hout, nullptr, nullptr, 0, pf(P0 + 5), nullptr,
         R_, 2 * H_, H_, H_, H_, 2 * H_, 0, 0, 0, 1, 0);
    gate_kernel<<<R_, 256, 0, stream>>>(hout, vec2, vfull, vfbf, xf, xbf);

    // ---- Stage B: geometric conv ----
    gemm(stream, xbf, WT_gc_wx, nullptr, nullptr, dxT, R_, nullptr, nullptr,
         R_, H_, H_, H_, H_, H_, 0, 0, 0, 1, 0);
    gemm(stream, xbf, WT_gc_wd, nullptr, nullptr, ddT, R_, nullptr, nullptr,
         R_, H_, H_, H_, H_, H_, 0, 0, 0, 1, 0);
    gemm(stream, vfbf, WT_gc_wv, nullptr, nullptr, vvT, 3 * R_, nullptr, nullptr,
         R_ * 3, H_, H_, H_, H_, H_, 0, 0, 0, 1, 0);
    buildw_kernel<<<dim3(N_ / 256, N_, B_), 256, 0, stream>>>(coords, w_bf, wdir);
    // agg_x = w @ dx (batched over B); BT = dxT with per-batch column offset
    gemm(stream, w_bf, dxT, aggx, nullptr, nullptr, 0, nullptr, nullptr,
         N_, H_, N_, N_, R_, H_, NN, 512, (long long)N_ * H_, B_, 0);
    // agg_v[c] = w @ vv[c] + (w*dirn[c]) @ dd   (channel-major layouts)
    for (int c = 0; c < 3; c++) {
        gemm(stream, w_bf, vvT + (size_t)c * R_, aggv + (size_t)c * R_ * H_,
             nullptr, nullptr, 0, nullptr, nullptr,
             N_, H_, N_, N_, 3 * R_, H_, NN, 512, (long long)N_ * H_, B_, 0);
        gemm(stream, wdir + (size_t)c * NN, ddT, aggv + (size_t)c * R_ * H_,
             nullptr, nullptr, 0, nullptr, aggv + (size_t)c * R_ * H_,
             N_, H_, N_, N_, R_, H_, 3 * NN, 512, (long long)N_ * H_, B_, 0);
    }
    postgc_kernel<<<R_, 256, 0, stream>>>(vfull, aggv, xf, aggx, normv, resid);
    gemm(stream, normv, WT_gc_wn, xf, xbf, nullptr, 0, nullptr, resid,
         R_, H_, H_, H_, H_, H_, 0, 0, 0, 1, 0);

    // ---- Stage C: transformer encoder layers ----
    for (int l = 0; l < NLAYERS_; l++) {
        const int base = LBASE + l * 16;
        gemm(stream, xbf, LW[l].wq, nullptr, qb, nullptr, 0, pf(base + 1), nullptr,
             R_, H_, H_, H_, H_, H_, 0, 0, 0, 1, 0);
        gemm(stream, xbf, LW[l].wk, nullptr, kb, nullptr, 0, pf(base + 3), nullptr,
             R_, H_, H_, H_, H_, H_, 0, 0, 0, 1, 0);
        gemm(stream, xbf, LW[l].wv, nullptr, vb2, nullptr, 0, pf(base + 5), nullptr,
             R_, H_, H_, H_, H_, H_, 0, 0, 0, 1, 0);
        attn_kernel<<<dim3(N_ / 128, NH_, B_), 256, 0, stream>>>(qb, kb, vb2, attf);
        cast_kernel<<<dim3((unsigned)(RH / 256)), 256, 0, stream>>>(attf, attbf, (int)RH);
        gemm(stream, attbf, LW[l].wo, projf, nullptr, nullptr, 0, pf(base + 7), xf,
             R_, H_, H_, H_, H_, H_, 0, 0, 0, 1, 0);
        ln_kernel<<<R_, 256, 0, stream>>>(projf, pf(base + 12), pf(base + 13), x1f, x1bf);
        gemm(stream, x1bf, LW[l].w1, nullptr, midbf, nullptr, 0, pf(base + 9), nullptr,
             R_, DFF_, H_, H_, H_, DFF_, 0, 0, 0, 1, 1 /*relu*/);
        gemm(stream, midbf, LW[l].w2, projf, nullptr, nullptr, 0, pf(base + 11), x1f,
             R_, H_, DFF_, DFF_, DFF_, H_, 0, 0, 0, 1, 0);
        ln_kernel<<<R_, 256, 0, stream>>>(projf, pf(base + 14), pf(base + 15), xf, xbf);
    }

    // ---- Stage D: pool + head ----
    pool_kernel<<<B_, 256, 0, stream>>>(xf, pooled);
    head_kernel<<<B_, 256, 0, stream>>>(pooled,
                                        pf(P0 + 10), pf(P0 + 11),
                                        pf(P0 + 12), pf(P0 + 13),
                                        pf(P0 + 14), pf(P0 + 15),
                                        (float*)d_out);
}
